// SimpleMoE_62672162783964
// MI455X (gfx1250) — matrix operand order repaired and checked by reading, not compiled, not run on hardware
//
#include <hip/hip_runtime.h>
#include <hip/hip_bf16.h>
#include <cmath>

// Problem constants (match reference)
#define NN   50000
#define EE   800000
#define INF_ 128
#define HIDF 64
#define OUTF 128
#define HDS  4
#define FCF  50
#define EPSBN 1e-5f
#define NSLOPE 0.2f

typedef __bf16 bf16_t;
typedef bf16_t v16bf __attribute__((ext_vector_type(16)));
typedef float  v8f   __attribute__((ext_vector_type(8)));

// ---- ordered-uint encoding of float for atomicMax-based segment max ----
__device__ __forceinline__ unsigned fenc(float f) {
  unsigned u = __float_as_uint(f);
  return (u & 0x80000000u) ? ~u : (u | 0x80000000u);
}
__device__ __forceinline__ float fdec(unsigned e) {
  unsigned u = (e & 0x80000000u) ? (e & 0x7FFFFFFFu) : ~e;
  return __uint_as_float(u);
}
#define ENC_NEG_INF 0x007FFFFFu  // fenc(-inf)

// ------------------------------ fills ---------------------------------
__global__ void fill_f32_k(float* __restrict__ p, float v, long n) {
  long i = (long)blockIdx.x * blockDim.x + threadIdx.x;
  if (i < n) p[i] = v;
}
__global__ void fill_u32_k(unsigned* __restrict__ p, unsigned v, long n) {
  long i = (long)blockIdx.x * blockDim.x + threadIdx.x;
  if (i < n) p[i] = v;
}

// --------------------- WMMA GEMM: C[N,M] = A[N,K] @ W[K,M] (+bias) ---------
// One wave32 per 16x16 tile. K and M templated so the K-loop fully unrolls
// into back-to-back v_wmma_f32_16x16x32_bf16 and every A/B/C address becomes
// base + compile-time immediate offset. bf16 operands, f32 accumulate.
template <int K, int M>
__global__ __launch_bounds__(32) void gemm_wmma_k(
    const float* __restrict__ A, const float* __restrict__ W,
    const float* __restrict__ bias, float* __restrict__ C)
{
  const int l   = threadIdx.x;      // 0..31
  const int mr  = l & 15;
  const int hi  = l >> 4;           // 0 or 1
  const long row0 = (long)blockIdx.x * 16;
  const int  col  = blockIdx.y * 16 + mr;
  const float* arow = A + (row0 + mr) * (long)K;
  const float* wcol = W + (long)hi * 16 * M + col;   // base for this lane's K-half

  v8f acc = {0.f,0.f,0.f,0.f,0.f,0.f,0.f,0.f};
#pragma unroll
  for (int k0 = 0; k0 < K; k0 += 32) {
    v16bf af, bfv;
#pragma unroll
    for (int j = 0; j < 16; ++j) {
      const int r = j >> 1, w = j & 1;
      // A k order used here: lanes0-15 hold K=0..15, lanes16-31 hold K=16..31 (matches B)
      const int ka = hi * 16 + 2 * r + w;
      af[j] = (bf16_t)arow[k0 + ka];
      // B k order used here (same as A): lanes0-15 hold K=0..15, lanes16-31 hold K=16..31
      bfv[j] = (bf16_t)wcol[(long)(k0 + j) * M];     // immediate offsets (M const)
    }
    acc = __builtin_amdgcn_wmma_f32_16x16x32_bf16(
        false, af, false, bfv, (short)0, acc, false, false);
  }
  // Hoist bias load + output base address out of the store loop.
  const float bv = bias ? bias[col] : 0.0f;
  float* crow = C + (row0 + 8 * hi) * (long)M + col;  // rows r+8*hi, r=0..7
#pragma unroll
  for (int r = 0; r < 8; ++r)
    crow[(long)r * M] = acc[r] + bv;
}

// ------------------------------ gate ----------------------------------
__global__ void gate_k(const float* __restrict__ x, const float* __restrict__ Wg,
                       const float* __restrict__ bg, float* __restrict__ gp, int n) {
  int i = blockIdx.x * blockDim.x + threadIdx.x;
  if (i >= n) return;
  const float* xr = x + (long)i * INF_;
  float z[4];
#pragma unroll
  for (int e = 0; e < 4; ++e) z[e] = bg[e];
  for (int c = 0; c < INF_; ++c) {
    float xv = xr[c];
#pragma unroll
    for (int e = 0; e < 4; ++e) z[e] += xv * Wg[c * 4 + e];
  }
  float m = fmaxf(fmaxf(z[0], z[1]), fmaxf(z[2], z[3]));
  float s = 0.f;
#pragma unroll
  for (int e = 0; e < 4; ++e) { z[e] = __expf(z[e] - m); s += z[e]; }
#pragma unroll
  for (int e = 0; e < 4; ++e) gp[(long)i * 4 + e] = z[e] / s;
}

// -------- GAT per-node attention coefficients: hs[n,h], hd[n,h] -------
__global__ void coeff_k(const float* __restrict__ h, const float* __restrict__ asrc,
                        const float* __restrict__ adst, float* __restrict__ hs,
                        float* __restrict__ hd, int n, int H, int C) {
  int i = blockIdx.x * blockDim.x + threadIdx.x;
  if (i >= n * H) return;
  int nd = i / H, hh = i % H;
  const float* hr = h + (long)nd * H * C + hh * C;
  const float* as = asrc + hh * C;
  const float* ad = adst + hh * C;
  float a = 0.f, b = 0.f;
  for (int c = 0; c < C; ++c) { a += hr[c] * as[c]; b += hr[c] * ad[c]; }
  hs[i] = a; hd[i] = b;
}

// ----- GAT alpha = leaky_relu(hs[s]+hd[d]) over E edges + N self-loops ----
__global__ void gat_alpha_max_k(const float* __restrict__ hs, const float* __restrict__ hd,
                                const int* __restrict__ src, const int* __restrict__ dst,
                                float* __restrict__ alpha, unsigned* __restrict__ mB,
                                int T, int E, int H) {
  int i = blockIdx.x * blockDim.x + threadIdx.x;
  if (i >= T * H) return;
  int t = i / H, hh = i % H;
  int s, d;
  if (t < E) { s = src[t]; d = dst[t]; } else { s = d = t - E; }
  float a = hs[s * H + hh] + hd[d * H + hh];
  a = a > 0.f ? a : NSLOPE * a;
  alpha[i] = a;
  atomicMax(&mB[d * H + hh], fenc(a));
}

// ----- Transformer alpha = dot(Q[d],K[s]) / sqrt(C); one wave32 per entry ----
__global__ void trans_alpha_max_k(const float* __restrict__ Q, const float* __restrict__ Kt,
                                  const int* __restrict__ src, const int* __restrict__ dst,
                                  float* __restrict__ alpha, unsigned* __restrict__ mB,
                                  int E, int H, int C, float scale) {
  long gid = (long)blockIdx.x * blockDim.x + threadIdx.x;
  int entry = (int)(gid >> 5);
  int lane  = (int)(gid & 31);
  if (entry >= E * H) return;
  int t = entry / H, hh = entry % H;
  int s = src[t], d = dst[t];
  const float* q = Q + (long)d * H * C + hh * C;
  const float* k = Kt + (long)s * H * C + hh * C;
  float acc = 0.f;
  for (int c = lane; c < C; c += 32) acc += q[c] * k[c];  // coalesced
#pragma unroll
  for (int o = 16; o > 0; o >>= 1) acc += __shfl_xor(acc, o, 32);
  if (lane == 0) {
    float a = acc * scale;
    alpha[entry] = a;
    atomicMax(&mB[d * H + hh], fenc(a));
  }
}

// ----- e = exp(a - max), scatter-sum of e per (dst,head) -----
__global__ void alpha_expsum_k(float* __restrict__ alpha, const unsigned* __restrict__ mB,
                               float* __restrict__ sB, const int* __restrict__ dst,
                               int T, int E, int H) {
  int i = blockIdx.x * blockDim.x + threadIdx.x;
  if (i >= T * H) return;
  int t = i / H, hh = i % H;
  int d = (t < E) ? dst[t] : (t - E);
  float m = fdec(mB[d * H + hh]);
  float e = __expf(alpha[i] - m);
  alpha[i] = e;
  atomicAdd(&sB[d * H + hh], e);
}

// ----- out[d,h,:] += (e/sum) * V[s,h,:]; one wave32 per (entry,head) ------
// Lanes stripe the channel dim -> coalesced gathers and coalesced atomics.
__global__ void scatter_agg_k(const float* __restrict__ alpha, const float* __restrict__ sB,
                              const int* __restrict__ src, const int* __restrict__ dst,
                              const float* __restrict__ V, float* __restrict__ out,
                              int T, int E, int H, int C) {
  long gid = (long)blockIdx.x * blockDim.x + threadIdx.x;
  int entry = (int)(gid >> 5);
  int lane  = (int)(gid & 31);
  if (entry >= T * H) return;
  int t = entry / H, hh = entry % H;
  int s, d;
  if (t < E) { s = src[t]; d = dst[t]; } else { s = d = t - E; }
  float w = alpha[entry] / (sB[d * H + hh] + 1e-16f);
  const float* vr = V + (long)s * H * C + hh * C;
  float* orow = out + (long)d * H * C + hh * C;
  for (int c = lane; c < C; c += 32) atomicAdd(&orow[c], w * vr[c]);
}

// ----- elementwise: out = act(a + b? + c? + bias[idx%M]?) -----
__global__ void ewise_k(const float* __restrict__ a, const float* __restrict__ b,
                        const float* __restrict__ c, const float* __restrict__ bias,
                        float* __restrict__ out, long n, int M, int relu) {
  long i = (long)blockIdx.x * blockDim.x + threadIdx.x;
  if (i >= n) return;
  float v = a[i];
  if (b) v += b[i];
  if (c) v += c[i];
  if (bias) v += bias[(int)(i % M)];
  if (relu) v = v > 0.f ? v : 0.f;
  out[i] = v;
}

// ----- BN affine prep: scale/shift per channel -----
__global__ void bn_prep_k(const float* __restrict__ g, const float* __restrict__ be,
                          const float* __restrict__ mu, const float* __restrict__ var,
                          float* __restrict__ scale, float* __restrict__ shift) {
  int c = threadIdx.x;
  if (c >= OUTF) return;
  float s = g[c] * rsqrtf(var[c] + EPSBN);
  scale[c] = s;
  shift[c] = be[c] - mu[c] * s;
}

// ----- BN + FC(128->50) -----
__global__ void bn_fc_k(const float* __restrict__ h, const float* __restrict__ scale,
                        const float* __restrict__ shift, const float* __restrict__ Wf,
                        const float* __restrict__ bf, float* __restrict__ out, int n) {
  int i = blockIdx.x * blockDim.x + threadIdx.x;
  if (i >= n * FCF) return;
  int nd = i / FCF, f = i % FCF;
  const float* hr = h + (long)nd * OUTF;
  float acc = bf[f];
  for (int c = 0; c < OUTF; ++c)
    acc += (hr[c] * scale[c] + shift[c]) * Wf[c * FCF + f];
  out[i] = acc;
}

// ----- gate-weighted combine of the 4 expert outputs -----
__global__ void combine_k(const float* __restrict__ eo, const float* __restrict__ gp,
                          float* __restrict__ out, int n) {
  int i = blockIdx.x * blockDim.x + threadIdx.x;
  if (i >= n * FCF) return;
  int nd = i / FCF;
  float acc = 0.f;
#pragma unroll
  for (int e = 0; e < 4; ++e)
    acc += gp[(long)nd * 4 + e] * eo[(long)e * n * FCF + i];
  out[i] = acc;
}

// ======================================================================
extern "C" void kernel_launch(void* const* d_in, const int* in_sizes, int n_in,
                              void* d_out, int out_size, void* d_ws, size_t ws_size,
                              hipStream_t stream) {
  (void)in_sizes; (void)n_in; (void)out_size; (void)ws_size;
  const int N = NN, E = EE;

  // ---- inputs in setup_inputs() dict (insertion) order, recursively ----
  int ip = 0;
  const float* x   = (const float*)d_in[ip++];
  const int*   ei  = (const int*)d_in[ip++];
  ip++; // batch (unused)
  const float* gW  = (const float*)d_in[ip++];
  const float* gb  = (const float*)d_in[ip++];
  // gat expert: W1, as1, ad1, b1, W2, as2, ad2, b2, g, be, mu, var, Wf, bf
  const float* gat[2][14];
  for (int e = 0; e < 2; ++e)
    for (int j = 0; j < 14; ++j) gat[e][j] = (const float*)d_in[ip++];
  // trans expert: q1.W q1.b k1.W k1.b v1.W v1.b s1.W s1.b
  //               q2.W q2.b k2.W k2.b v2.W v2.b s2.W s2.b res.W res.b
  //               g be mu var Wf bf
  const float* tr[2][24];
  for (int e = 0; e < 2; ++e)
    for (int j = 0; j < 24; ++j) tr[e][j] = (const float*)d_in[ip++];

  const int* src = ei;
  const int* dst = ei + E;

  // ---- workspace layout ----
  float* ws = (float*)d_ws;
  size_t off = 0;
  auto alloc = [&](size_t cnt) { float* p = ws + off; off += cnt; return p; };
  float* gateP  = alloc((size_t)N * 4);
  float* B0 = alloc((size_t)N * 256);
  float* B1 = alloc((size_t)N * 256);
  float* B2 = alloc((size_t)N * 256);
  float* B3 = alloc((size_t)N * 256);
  float* B4 = alloc((size_t)N * 256);
  float* Br = alloc((size_t)N * 128);
  float* hsB = alloc((size_t)N * 4);
  float* hdB = alloc((size_t)N * 4);
  float* alphaB = alloc((size_t)(E + N) * 4);
  unsigned* mB = (unsigned*)alloc((size_t)N * 4);
  float* sB = alloc((size_t)N * 4);
  float* scaleB = alloc(256);
  float* shiftB = alloc(256);
  float* eo = alloc((size_t)4 * N * FCF);

  auto cdiv = [](long a, long b) { return (int)((a + b - 1) / b); };

  auto gemm = [&](const float* A, const float* W, const float* bias, float* C,
                  int K, int M) {
    dim3 grid(N / 16, M / 16);
    if (K == 128 && M == 256)
      gemm_wmma_k<128, 256><<<grid, 32, 0, stream>>>(A, W, bias, C);
    else if (K == 256 && M == 128)
      gemm_wmma_k<256, 128><<<grid, 32, 0, stream>>>(A, W, bias, C);
    else  // K==128, M==128
      gemm_wmma_k<128, 128><<<grid, 32, 0, stream>>>(A, W, bias, C);
  };

  auto gatconv = [&](const float* hmat, int H, int C, const float* asrc,
                     const float* adst, float* agg) {
    long NH = (long)N * H;
    coeff_k<<<cdiv(NH, 256), 256, 0, stream>>>(hmat, asrc, adst, hsB, hdB, N, H, C);
    fill_u32_k<<<cdiv(NH, 256), 256, 0, stream>>>(mB, ENC_NEG_INF, NH);
    fill_f32_k<<<cdiv(NH, 256), 256, 0, stream>>>(sB, 0.f, NH);
    const int T = E + N;  // self-loops appended
    gat_alpha_max_k<<<cdiv((long)T * H, 256), 256, 0, stream>>>(
        hsB, hdB, src, dst, alphaB, mB, T, E, H);
    alpha_expsum_k<<<cdiv((long)T * H, 256), 256, 0, stream>>>(
        alphaB, mB, sB, dst, T, E, H);
    fill_f32_k<<<cdiv((long)N * H * C, 256), 256, 0, stream>>>(agg, 0.f, (long)N * H * C);
    scatter_agg_k<<<cdiv((long)T * H * 32, 256), 256, 0, stream>>>(
        alphaB, sB, src, dst, hmat, agg, T, E, H, C);
  };

  auto transconv = [&](const float* Q, const float* Kt, const float* V, int H,
                       int C, float* agg) {
    long NH = (long)N * H;
    fill_u32_k<<<cdiv(NH, 256), 256, 0, stream>>>(mB, ENC_NEG_INF, NH);
    fill_f32_k<<<cdiv(NH, 256), 256, 0, stream>>>(sB, 0.f, NH);
    float scale = 1.0f / sqrtf((float)C);
    trans_alpha_max_k<<<cdiv((long)E * H * 32, 256), 256, 0, stream>>>(
        Q, Kt, src, dst, alphaB, mB, E, H, C, scale);
    alpha_expsum_k<<<cdiv((long)E * H, 256), 256, 0, stream>>>(
        alphaB, mB, sB, dst, E, E, H);
    fill_f32_k<<<cdiv((long)N * H * C, 256), 256, 0, stream>>>(agg, 0.f, (long)N * H * C);
    scatter_agg_k<<<cdiv((long)E * H * 32, 256), 256, 0, stream>>>(
        alphaB, sB, src, dst, V, agg, E, E, H, C);
  };

  // ---- gate ----
  gate_k<<<cdiv(N, 256), 256, 0, stream>>>(x, gW, gb, gateP, N);

  // ---- GAT experts (outputs -> eo slots 0,1) ----
  for (int e = 0; e < 2; ++e) {
    const float** p = gat[e];
    // layer 1: h = relu(gatconv(x) + b1), heads=4, ch=64, concat
    gemm(x, p[0], nullptr, B0, INF_, HDS * HIDF);            // x @ W1 -> [N,256]
    gatconv(B0, HDS, HIDF, p[1], p[2], B1);
    ewise_k<<<cdiv((long)N * 256, 256), 256, 0, stream>>>(
        B1, nullptr, nullptr, p[3], B1, (long)N * 256, 256, 1);
    // layer 2: h = relu(gatconv(h) + b2), heads=1, ch=128
    gemm(B1, p[4], nullptr, B0, 256, OUTF);                  // h1 @ W2 -> [N,128]
    gatconv(B0, 1, OUTF, p[5], p[6], B2);
    ewise_k<<<cdiv((long)N * OUTF, 256), 256, 0, stream>>>(
        B2, nullptr, nullptr, p[7], B2, (long)N * OUTF, OUTF, 1);
    // BN + FC
    bn_prep_k<<<1, OUTF, 0, stream>>>(p[8], p[9], p[10], p[11], scaleB, shiftB);
    bn_fc_k<<<cdiv((long)N * FCF, 256), 256, 0, stream>>>(
        B2, scaleB, shiftB, p[12], p[13], eo + (size_t)e * N * FCF, N);
  }

  // ---- Transformer experts (outputs -> eo slots 2,3) ----
  for (int e = 0; e < 2; ++e) {
    const float** p = tr[e];
    gemm(x, p[16], p[17], Br, INF_, OUTF);                   // res = x@Wres+b
    gemm(x, p[0], p[1], B0, INF_, 256);                      // Q1
    gemm(x, p[2], p[3], B1, INF_, 256);                      // K1
    gemm(x, p[4], p[5], B2, INF_, 256);                      // V1
    gemm(x, p[6], p[7], B3, INF_, 256);                      // S1 (skip)
    transconv(B0, B1, B2, HDS, HIDF, B4);
    ewise_k<<<cdiv((long)N * 256, 256), 256, 0, stream>>>(   // h1 = relu(agg + S1)
        B4, B3, nullptr, nullptr, B4, (long)N * 256, 256, 1);
    gemm(B4, p[8],  p[9],  B0, 256, OUTF);                   // Q2
    gemm(B4, p[10], p[11], B1, 256, OUTF);                   // K2
    gemm(B4, p[12], p[13], B2, 256, OUTF);                   // V2
    gemm(B4, p[14], p[15], B3, 256, OUTF);                   // S2
    transconv(B0, B1, B2, 1, OUTF, B4);
    ewise_k<<<cdiv((long)N * OUTF, 256), 256, 0, stream>>>(  // relu(agg + S2 + res)
        B4, B3, Br, nullptr, B4, (long)N * OUTF, OUTF, 1);
    bn_prep_k<<<1, OUTF, 0, stream>>>(p[18], p[19], p[20], p[21], scaleB, shiftB);
    bn_fc_k<<<cdiv((long)N * FCF, 256), 256, 0, stream>>>(
        B4, scaleB, shiftB, p[22], p[23], eo + (size_t)(2 + e) * N * FCF, N);
  }

  // ---- gate-weighted combine ----
  combine_k<<<cdiv((long)N * FCF, 256), 256, 0, stream>>>(eo, gateP, (float*)d_out, N);
}
